// TransformerLayer_69965017252156
// MI455X (gfx1250) — matrix operand order repaired and checked by reading, not compiled, not run on hardware
//
#include <hip/hip_runtime.h>
#include <math.h>

// ---------------------------------------------------------------------------
// MI455X (gfx1250) transformer layer: LN -> attention -> +res -> LN -> MLP -> +res
// Matmuls: v_wmma_f32_16x16x32_bf16 (wave32 WMMA).
// GEMM uses async global->LDS staging (ASYNCcnt) with double buffering.
// ---------------------------------------------------------------------------

typedef __bf16 bf16_t;
typedef __attribute__((ext_vector_type(16))) __bf16 v16bf;
typedef __attribute__((ext_vector_type(8)))  __bf16 v8bf;
typedef __attribute__((ext_vector_type(8)))  float  v8f;

#define SEQ    2048
#define BATCH  2
#define DMODEL 1024
#define DFF    4096
#define NH     16
#define HD     64
#define MTOT   (BATCH*SEQ)   // 4096 rows

__device__ __forceinline__ v16bf cat8(v8bf a, v8bf b) {
  return __builtin_shufflevector(a, b, 0,1,2,3,4,5,6,7,8,9,10,11,12,13,14,15);
}

__device__ __forceinline__ float half16_max(float v) {
  v = fmaxf(v, __shfl_xor(v, 1, 32));
  v = fmaxf(v, __shfl_xor(v, 2, 32));
  v = fmaxf(v, __shfl_xor(v, 4, 32));
  v = fmaxf(v, __shfl_xor(v, 8, 32));
  return v;
}
__device__ __forceinline__ float half16_sum(float v) {
  v += __shfl_xor(v, 1, 32);
  v += __shfl_xor(v, 2, 32);
  v += __shfl_xor(v, 4, 32);
  v += __shfl_xor(v, 8, 32);
  return v;
}

// Async global->LDS copy of 16 bytes (CDNA5, tracked by ASYNCcnt).
// ldsdst: 32-bit LDS byte offset; gsrc: 64-bit global address.
__device__ __forceinline__ void async_copy16(unsigned ldsdst, const void* gsrc) {
  unsigned long long ga = (unsigned long long)(size_t)gsrc;
  asm volatile("global_load_async_to_lds_b128 %0, %1, off"
               :: "v"(ldsdst), "v"(ga) : "memory");
}
__device__ __forceinline__ void wait_async0() {
  asm volatile("s_wait_asynccnt 0x0" ::: "memory");
}

// ---------------------------------------------------------------------------
// f32 -> bf16 cast (grid-strided)
// ---------------------------------------------------------------------------
__global__ void cast_bf16_kernel(const float* __restrict__ in,
                                 bf16_t* __restrict__ out, int n) {
  int i = blockIdx.x * blockDim.x + threadIdx.x;
  int stride = gridDim.x * blockDim.x;
  for (; i < n; i += stride) out[i] = (bf16_t)in[i];
}

// ---------------------------------------------------------------------------
// LayerNorm over last dim (1024), unbiased variance (ddof=1), bf16 out
// ---------------------------------------------------------------------------
__global__ void ln_kernel(const float* __restrict__ x,
                          const float* __restrict__ mean_scale,
                          const float* __restrict__ std_scale,
                          bf16_t* __restrict__ out) {
  const int row = blockIdx.x;
  const int t = threadIdx.x;  // 256
  const float* xr = x + (size_t)row * DMODEL;
  float v[4];
  float s = 0.f, s2 = 0.f;
#pragma unroll
  for (int j = 0; j < 4; ++j) {
    v[j] = xr[t + j * 256];
    s  += v[j];
    s2 += v[j] * v[j];
  }
  __shared__ float red[256], red2[256];
  red[t] = s; red2[t] = s2;
  __syncthreads();
  for (int off = 128; off > 0; off >>= 1) {
    if (t < off) { red[t] += red[t + off]; red2[t] += red2[t + off]; }
    __syncthreads();
  }
  const float mean = red[0] * (1.0f / DMODEL);
  const float var  = (red2[0] - (float)DMODEL * mean * mean) * (1.0f / (DMODEL - 1));
  const float inv  = rsqrtf(var + 1e-9f);
#pragma unroll
  for (int j = 0; j < 4; ++j) {
    const int d = t + j * 256;
    out[(size_t)row * DMODEL + d] =
        (bf16_t)((v[j] - mean) * inv * std_scale[d] + mean_scale[d]);
  }
}

// ---------------------------------------------------------------------------
// Y[M,N] = act( A[M,K] @ B[N,K]^T + resid )   (nn.Linear layout: B rows = K)
// block = 256 threads = 8 waves (2 M x 4 N), block tile 128x128.
// K stepped by 32; A/B tiles staged global->LDS with async copies,
// double-buffered (s_wait_asynccnt + barrier), fragments read via ds_load.
// ---------------------------------------------------------------------------
__global__ __launch_bounds__(256, 1)
void gemm_kernel(const bf16_t* __restrict__ A, const bf16_t* __restrict__ B,
                 const float* __restrict__ resid, float* __restrict__ outF,
                 bf16_t* __restrict__ outH, int M, int N, int K, int act) {
  __shared__ alignas(16) bf16_t Atile[2][128 * 32];
  __shared__ alignas(16) bf16_t Btile[2][128 * 32];
  const int t = threadIdx.x;
  const int lane = t & 31;
  const int wid  = t >> 5;
  const int m15  = lane & 15;
  const int half = lane >> 4;
  const int waveM = wid & 1;
  const int waveN = wid >> 1;
  const int rowA0 = blockIdx.y * 128;
  const int colB0 = blockIdx.x * 128;

  // Staging: 128x32 bf16 tile = 4096 elems; 256 threads x 16 elems (2 x b128).
  const int flat = t * 16;
  const int srow = flat >> 5;     // 0..127
  const int skk  = flat & 31;     // 0 or 16
  const bf16_t* Asrc = A + (size_t)(rowA0 + srow) * K + skk;
  const bf16_t* Bsrc = B + (size_t)(colB0 + srow) * K + skk;
  const unsigned ldsAbase = (unsigned)(size_t)&Atile[0][0];
  const unsigned ldsBbase = (unsigned)(size_t)&Btile[0][0];
  const unsigned soff = (unsigned)((srow * 32 + skk) * 2);

  v8f acc[4][2];
#pragma unroll
  for (int mi = 0; mi < 4; ++mi)
#pragma unroll
    for (int ni = 0; ni < 2; ++ni)
#pragma unroll
      for (int i = 0; i < 8; ++i) acc[mi][ni][i] = 0.f;

  // prime buffer 0
  async_copy16(ldsAbase + soff,      Asrc);
  async_copy16(ldsAbase + soff + 16, Asrc + 8);
  async_copy16(ldsBbase + soff,      Bsrc);
  async_copy16(ldsBbase + soff + 16, Bsrc + 8);
  wait_async0();
  __syncthreads();

  int cur = 0;
  for (int k0 = 0; k0 < K; k0 += 32) {
    const int nxt = cur ^ 1;
    if (k0 + 32 < K) {
      const unsigned aoff = ldsAbase + (unsigned)(nxt * 8192) + soff;
      const unsigned boff = ldsBbase + (unsigned)(nxt * 8192) + soff;
      async_copy16(aoff,      Asrc + k0 + 32);
      async_copy16(aoff + 16, Asrc + k0 + 40);
      async_copy16(boff,      Bsrc + k0 + 32);
      async_copy16(boff + 16, Bsrc + k0 + 40);
    }

    const bf16_t* As = &Atile[cur][0];
    const bf16_t* Bs = &Btile[cur][0];
    v16bf af[4], bfr[2];
#pragma unroll
    for (int mi = 0; mi < 4; ++mi) {
      const bf16_t* p = As + (waveM * 64 + mi * 16 + m15) * 32;
      af[mi] = *(const v16bf*)(p + half * 16);
    }
#pragma unroll
    for (int ni = 0; ni < 2; ++ni) {
      bfr[ni] = *(const v16bf*)(Bs + (waveN * 32 + ni * 16 + m15) * 32 + half * 16);
    }
#pragma unroll
    for (int mi = 0; mi < 4; ++mi)
#pragma unroll
      for (int ni = 0; ni < 2; ++ni)
        acc[mi][ni] = __builtin_amdgcn_wmma_f32_16x16x32_bf16(
            false, af[mi], false, bfr[ni], (short)0, acc[mi][ni], false, false);

    wait_async0();      // next tile landed in LDS
    __syncthreads();    // everyone done reading cur, everyone's fill visible
    cur = nxt;
  }

#pragma unroll
  for (int mi = 0; mi < 4; ++mi)
#pragma unroll
    for (int ni = 0; ni < 2; ++ni)
#pragma unroll
      for (int i = 0; i < 8; ++i) {
        const int row = rowA0 + waveM * 64 + mi * 16 + i + half * 8;
        const int col = colB0 + waveN * 32 + ni * 16 + m15;
        float v = acc[mi][ni][i];
        if (resid) v += resid[(size_t)row * N + col];
        if (act)   v = 0.5f * v * (1.0f + erff(v * 0.70710678f));  // exact GELU
        if (outF) outF[(size_t)row * N + col] = v;
        if (outH) outH[(size_t)row * N + col] = (bf16_t)v;
      }
}

// ---------------------------------------------------------------------------
// V[b,s,h,hd] (stored as [4096,1024]) -> Vt[b,h,hd,s]
// ---------------------------------------------------------------------------
__global__ void vtrans_kernel(const bf16_t* __restrict__ vin,
                              bf16_t* __restrict__ vt) {
  size_t idx = (size_t)blockIdx.x * 256 + threadIdx.x;  // over 4M elements
  const int sidx = (int)(idx & (SEQ - 1));
  size_t r = idx >> 11;
  const int d = (int)(r & (HD - 1));
  size_t bh = r >> 6;
  const int h = (int)(bh & (NH - 1));
  const int b = (int)(bh >> 4);
  vt[idx] = vin[((size_t)(b * SEQ + sidx)) * DMODEL + h * HD + d];
}

// ---------------------------------------------------------------------------
// Causal flash attention: one wave per 16-query tile per (b,h).
// Q,K in [b*S, D] bf16 (head offset h*64), Vt in [b,h,d,s] bf16.
// P (scores) converted C-layout -> A-fragment layout via per-wave LDS tile.
// ---------------------------------------------------------------------------
__global__ __launch_bounds__(256, 1)
void attn_kernel(const bf16_t* __restrict__ Q, const bf16_t* __restrict__ K,
                 const bf16_t* __restrict__ Vt, bf16_t* __restrict__ preout) {
  __shared__ alignas(16) bf16_t plds[8][16 * 32];  // per-wave P tile
  const int lane = threadIdx.x & 31;
  const int wid  = threadIdx.x >> 5;
  const int m15  = lane & 15;
  const int half = lane >> 4;
  const int qtile = blockIdx.x * 8 + wid;   // 0..127
  const int h = blockIdx.y;
  const int b = blockIdx.z;
  const int q0 = qtile * 16;

  const bf16_t* Qb = Q + (size_t)b * SEQ * DMODEL + (size_t)h * HD;
  const bf16_t* Kb = K + (size_t)b * SEQ * DMODEL + (size_t)h * HD;
  const bf16_t* Vb = Vt + ((size_t)(b * NH + h) * HD) * SEQ;
  bf16_t* pl = &plds[wid][0];

  v16bf qf[2];
#pragma unroll
  for (int c = 0; c < 2; ++c) {
    const bf16_t* p = Qb + (size_t)(q0 + m15) * DMODEL + c * 32;
    qf[c] = *(const v16bf*)(p + half * 16);
  }

  v8f o[4];
  float mrun[8], lrun[8];
#pragma unroll
  for (int i = 0; i < 8; ++i) {
    mrun[i] = -__builtin_inff();
    lrun[i] = 0.f;
    o[0][i] = 0.f; o[1][i] = 0.f; o[2][i] = 0.f; o[3][i] = 0.f;
  }

  const int kbend = (q0 + 47) >> 5;  // 32-key blocks for the causal window
  for (int kb = 0; kb < kbend; ++kb) {
    const int k0 = kb * 32;
    v8f s0, s1;
#pragma unroll
    for (int i = 0; i < 8; ++i) { s0[i] = 0.f; s1[i] = 0.f; }
#pragma unroll
    for (int c = 0; c < 2; ++c) {
      v16bf kf0 = *(const v16bf*)(Kb + (size_t)(k0 + m15) * DMODEL + c * 32 + half * 16);
      v16bf kf1 = *(const v16bf*)(Kb + (size_t)(k0 + 16 + m15) * DMODEL + c * 32 + half * 16);
      s0 = __builtin_amdgcn_wmma_f32_16x16x32_bf16(false, qf[c], false, kf0, (short)0, s0, false, false);
      s1 = __builtin_amdgcn_wmma_f32_16x16x32_bf16(false, qf[c], false, kf1, (short)0, s1, false, false);
    }
    // online softmax (rows live in 16-lane halves: row = i + half*8)
#pragma unroll
    for (int i = 0; i < 8; ++i) {
      const int qrow = q0 + i + half * 8;
      float e0 = s0[i] * 0.125f;   // 1/sqrt(64)
      float e1 = s1[i] * 0.125f;
      if (k0 + m15 > qrow)      e0 = -__builtin_inff();
      if (k0 + 16 + m15 > qrow) e1 = -__builtin_inff();
      const float mb   = half16_max(fmaxf(e0, e1));
      const float mnew = fmaxf(mrun[i], mb);
      const float sc   = __expf(mrun[i] - mnew);
      const float p0   = __expf(e0 - mnew);
      const float p1   = __expf(e1 - mnew);
      const float rs   = half16_sum(p0 + p1);
      lrun[i] = lrun[i] * sc + rs;
      mrun[i] = mnew;
      o[0][i] *= sc; o[1][i] *= sc; o[2][i] *= sc; o[3][i] *= sc;
      const int row = i + half * 8;
      pl[row * 32 + m15]      = (bf16_t)p0;
      pl[row * 32 + 16 + m15] = (bf16_t)p1;
    }
    // P: C-layout -> A-fragment via LDS (wave-private region, DS ops in-order)
    const bf16_t* pp = pl + m15 * 32;
    v16bf pf = *(const v16bf*)(pp + half * 16);
#pragma unroll
    for (int tt = 0; tt < 4; ++tt) {
      v16bf vf = *(const v16bf*)(Vb + (size_t)(tt * 16 + m15) * SEQ + k0 + half * 16);
      o[tt] = __builtin_amdgcn_wmma_f32_16x16x32_bf16(false, pf, false, vf, (short)0, o[tt], false, false);
    }
  }

#pragma unroll
  for (int i = 0; i < 8; ++i) {
    const float inv = 1.0f / lrun[i];
    o[0][i] *= inv; o[1][i] *= inv; o[2][i] *= inv; o[3][i] *= inv;
  }
#pragma unroll
  for (int tt = 0; tt < 4; ++tt)
#pragma unroll
    for (int i = 0; i < 8; ++i) {
      const int row = q0 + i + half * 8;
      preout[((size_t)(b * SEQ + row)) * DMODEL + h * HD + tt * 16 + m15] = (bf16_t)o[tt][i];
    }
}

// ---------------------------------------------------------------------------
// Host orchestration
// ---------------------------------------------------------------------------
extern "C" void kernel_launch(void* const* d_in, const int* in_sizes, int n_in,
                              void* d_out, int out_size, void* d_ws, size_t ws_size,
                              hipStream_t stream) {
  (void)in_sizes; (void)n_in; (void)out_size; (void)ws_size;
  const float* x    = (const float*)d_in[0];
  const float* Wq   = (const float*)d_in[1];
  const float* Wk   = (const float*)d_in[2];
  const float* Wv   = (const float*)d_in[3];
  const float* Wo   = (const float*)d_in[4];
  const float* Wup  = (const float*)d_in[5];
  const float* Wdn  = (const float*)d_in[6];
  const float* ln1m = (const float*)d_in[7];
  const float* ln1s = (const float*)d_in[8];
  const float* ln2m = (const float*)d_in[9];
  const float* ln2s = (const float*)d_in[10];
  float* out = (float*)d_out;

  char* ws = (char*)d_ws;
  size_t off = 0;
  auto carve = [&](size_t bytes) -> char* {
    char* p = ws + off;
    off += (bytes + 255) & ~(size_t)255;
    return p;
  };
  bf16_t* ln1  = (bf16_t*)carve((size_t)MTOT * DMODEL * 2);
  bf16_t* ln2  = (bf16_t*)carve((size_t)MTOT * DMODEL * 2);
  bf16_t* wqh  = (bf16_t*)carve((size_t)DMODEL * DMODEL * 2);
  bf16_t* wkh  = (bf16_t*)carve((size_t)DMODEL * DMODEL * 2);
  bf16_t* wvh  = (bf16_t*)carve((size_t)DMODEL * DMODEL * 2);
  bf16_t* woh  = (bf16_t*)carve((size_t)DMODEL * DMODEL * 2);
  bf16_t* wuph = (bf16_t*)carve((size_t)DFF * DMODEL * 2);
  bf16_t* wdnh = (bf16_t*)carve((size_t)DMODEL * DFF * 2);
  bf16_t* qh   = (bf16_t*)carve((size_t)MTOT * DMODEL * 2);
  bf16_t* kh   = (bf16_t*)carve((size_t)MTOT * DMODEL * 2);
  bf16_t* vh   = (bf16_t*)carve((size_t)MTOT * DMODEL * 2);
  bf16_t* vt   = (bf16_t*)carve((size_t)MTOT * DMODEL * 2);
  bf16_t* pre  = (bf16_t*)carve((size_t)MTOT * DMODEL * 2);
  float*  x1   = (float*)carve((size_t)MTOT * DMODEL * 4);
  bf16_t* mid  = (bf16_t*)carve((size_t)MTOT * DFF * 2);

  // 1. weights -> bf16
  cast_bf16_kernel<<<512, 256, 0, stream>>>(Wq,  wqh,  DMODEL * DMODEL);
  cast_bf16_kernel<<<512, 256, 0, stream>>>(Wk,  wkh,  DMODEL * DMODEL);
  cast_bf16_kernel<<<512, 256, 0, stream>>>(Wv,  wvh,  DMODEL * DMODEL);
  cast_bf16_kernel<<<512, 256, 0, stream>>>(Wo,  woh,  DMODEL * DMODEL);
  cast_bf16_kernel<<<512, 256, 0, stream>>>(Wup, wuph, DFF * DMODEL);
  cast_bf16_kernel<<<512, 256, 0, stream>>>(Wdn, wdnh, DFF * DMODEL);

  // 2. LN1
  ln_kernel<<<MTOT, 256, 0, stream>>>(x, ln1m, ln1s, ln1);

  // 3. Q/K/V projections (WMMA, async-LDS staged)
  gemm_kernel<<<dim3(DMODEL / 128, MTOT / 128), 256, 0, stream>>>(
      ln1, wqh, nullptr, nullptr, qh, MTOT, DMODEL, DMODEL, 0);
  gemm_kernel<<<dim3(DMODEL / 128, MTOT / 128), 256, 0, stream>>>(
      ln1, wkh, nullptr, nullptr, kh, MTOT, DMODEL, DMODEL, 0);
  gemm_kernel<<<dim3(DMODEL / 128, MTOT / 128), 256, 0, stream>>>(
      ln1, wvh, nullptr, nullptr, vh, MTOT, DMODEL, DMODEL, 0);

  // 4. V -> V^T per head
  vtrans_kernel<<<(MTOT * DMODEL) / 256, 256, 0, stream>>>(vh, vt);

  // 5. causal flash attention (WMMA)
  attn_kernel<<<dim3(SEQ / (16 * 8), NH, BATCH), 256, 0, stream>>>(qh, kh, vt, pre);

  // 6. output projection + residual
  gemm_kernel<<<dim3(DMODEL / 128, MTOT / 128), 256, 0, stream>>>(
      pre, woh, x, x1, nullptr, MTOT, DMODEL, DMODEL, 0);

  // 7. LN2
  ln_kernel<<<MTOT, 256, 0, stream>>>(x1, ln2m, ln2s, ln2);

  // 8. MLP up + exact GELU
  gemm_kernel<<<dim3(DFF / 128, MTOT / 128), 256, 0, stream>>>(
      ln2, wuph, nullptr, nullptr, mid, MTOT, DFF, DMODEL, 1);

  // 9. MLP down + residual -> output
  gemm_kernel<<<dim3(DMODEL / 128, MTOT / 128), 256, 0, stream>>>(
      mid, wdnh, x1, out, nullptr, MTOT, DMODEL, DFF, 0);
}
